// JointAttention_44890998178481
// MI455X (gfx1250) — matrix operand order repaired and checked by reading, not compiled, not run on hardware
//
#include <hip/hip_runtime.h>

// ---------------------------------------------------------------------------
// CDNA5 (gfx1250) bf16-WMMA JointAttention.
//   - wave32 blocks, v_wmma_f32_16x16x32_bf16 (f32 accumulate) everywhere
//   - fragment-friendly layouts: all A/B fragments are 128-bit loads
//   - attention: K row-major + V pre-transposed => K/V fragments load
//     directly from global, only the P transpose goes through LDS
//   - vectorized epilogues where the 8 rows a lane owns are contiguous
//     in the destination (t / st branches; L % 16 == 0)
// ---------------------------------------------------------------------------

typedef __attribute__((ext_vector_type(16))) __bf16 v16bf;
typedef __attribute__((ext_vector_type(8)))  __bf16 bf16x8;
typedef __attribute__((ext_vector_type(4)))  __bf16 bf16x4;
typedef __attribute__((ext_vector_type(2)))  __bf16 bf16x2;
typedef __attribute__((ext_vector_type(8)))  float  v8f;

#define PC_     128
#define HEADS_  8
#define HD_     16
#define B_      8
#define N_      25
#define T_      64
#define C_      512
#define TOKENS_ 12800   // B*N*T — same token count for every branch

__device__ __forceinline__ v8f wmma_bf16(v16bf a, v16bf b, v8f c) {
    return __builtin_amdgcn_wmma_f32_16x16x32_bf16(false, a, false, b,
                                                   (short)0, c, false, false);
}
__device__ __forceinline__ v16bf cat8(bf16x8 lo, bf16x8 hi) {
    return __builtin_shufflevector(lo, hi, 0,1,2,3,4,5,6,7,8,9,10,11,12,13,14,15);
}

// --------------------------- branch traits --------------------------------
template <int BR> struct Branch;
template <> struct Branch<0> {   // 't': (b*n) seqs of length t
    static constexpr int L = 64, Lp = 64, chOff = 384, g = 1, SEQS = 200, QB = 2;
    static constexpr bool CONTIG = true;   // L % 16 == 0
    __device__ static void tok2bnt(int tau, int& b, int& n, int& t) {
        int seq = tau >> 6; int j = tau & 63;
        b = seq / 25; n = seq - b * 25; t = j;
    }
};
template <> struct Branch<1> {   // 's': (b*t) seqs of length n
    static constexpr int L = 25, Lp = 32, chOff = 256, g = 2, SEQS = 512, QB = 1;
    static constexpr bool CONTIG = false;
    __device__ static void tok2bnt(int tau, int& b, int& n, int& t) {
        int seq = tau / 25; int j = tau - seq * 25;
        b = seq >> 6; t = seq & 63; n = j;
    }
};
template <> struct Branch<2> {   // 'st': b seqs of length n*t
    static constexpr int L = 1600, Lp = 1600, chOff = 128, g = 3, SEQS = 8, QB = 50;
    static constexpr bool CONTIG = true;
    __device__ static void tok2bnt(int tau, int& b, int& n, int& t) {
        int seq = tau / 1600; int j = tau - seq * 1600;
        b = seq; n = j >> 6; t = j & 63;
    }
};

// ------------------------ phase 1: QKV projection -------------------------
// grid = (TOKENS/128, 24), 1 wave. Stage one transposed weight tile, hoist
// B-fragments to registers, then 8 token tiles of [16x16], K=128.
// nt 0..7 -> Q heads, 8..15 -> K heads, 16..23 -> V heads (V transposed).
template <int BR>
__global__ __launch_bounds__(32)
void qkv_kernel(const float* __restrict__ x, const float* __restrict__ wqkv,
                const float* __restrict__ bqkv,
                __bf16* __restrict__ qbuf, __bf16* __restrict__ kbuf,
                __bf16* __restrict__ vT)
{
    __shared__ __bf16 BtT[16][128];   // weight tile transposed: [col][k]
    const int lane = threadIdx.x;
    const int row = lane & 15, grp = lane >> 4, col = lane & 15;
    const int nt = blockIdx.y;

    // stage BtT: lane owns 4 consecutive k-rows -> packed b64 stores
    {
        const int k0 = lane * 4;
        float w[4][16];
        #pragma unroll
        for (int q = 0; q < 4; ++q) {
            const float4* src = (const float4*)(wqkv + (size_t)(k0 + q) * 384 + nt * 16);
            #pragma unroll
            for (int c4 = 0; c4 < 4; ++c4) {
                float4 v = src[c4];
                w[q][c4*4+0] = v.x; w[q][c4*4+1] = v.y;
                w[q][c4*4+2] = v.z; w[q][c4*4+3] = v.w;
            }
        }
        #pragma unroll
        for (int c = 0; c < 16; ++c) {
            bf16x4 p = { (__bf16)w[0][c], (__bf16)w[1][c],
                         (__bf16)w[2][c], (__bf16)w[3][c] };
            *(bf16x4*)&BtT[c][k0] = p;
        }
    }
    __syncthreads();

    // B-fragments in registers (reused by all 8 token tiles)
    v16bf bm[4];
    #pragma unroll
    for (int ks = 0; ks < 4; ++ks) {
        bf16x8 lo = *(const bf16x8*)&BtT[col][ks*32 + grp*8];
        bf16x8 hi = *(const bf16x8*)&BtT[col][ks*32 + 16 + grp*8];
        bm[ks] = cat8(lo, hi);
    }

    const float bias = bqkv[nt*16 + col];
    const int sec = nt >> 3, h = nt & 7;

    for (int it = 0; it < 8; ++it) {
        const int tt = blockIdx.x * 8 + it;
        int b, n, t;
        Branch<BR>::tok2bnt(tt*16 + row, b, n, t);
        const float* arow = x + ((size_t)(b*N_ + n)*T_ + t)*C_ + Branch<BR>::chOff;

        v8f acc;
        #pragma unroll
        for (int r = 0; r < 8; ++r) acc[r] = bias;

        #pragma unroll
        for (int ks = 0; ks < 4; ++ks) {
            // A-fragment: two contiguous 8-float runs of this row
            float4 f0 = *(const float4*)(arow + ks*32 + grp*8);
            float4 f1 = *(const float4*)(arow + ks*32 + grp*8 + 4);
            float4 f2 = *(const float4*)(arow + ks*32 + 16 + grp*8);
            float4 f3 = *(const float4*)(arow + ks*32 + 16 + grp*8 + 4);
            v16bf a = { (__bf16)f0.x,(__bf16)f0.y,(__bf16)f0.z,(__bf16)f0.w,
                        (__bf16)f1.x,(__bf16)f1.y,(__bf16)f1.z,(__bf16)f1.w,
                        (__bf16)f2.x,(__bf16)f2.y,(__bf16)f2.z,(__bf16)f2.w,
                        (__bf16)f3.x,(__bf16)f3.y,(__bf16)f3.z,(__bf16)f3.w };
            acc = wmma_bf16(a, bm[ks], acc);
        }

        if (sec == 2) {
            // V transposed: [seq][h][d][j]; the 8 rows this lane owns are
            // 8 consecutive j positions when L % 16 == 0 -> one b128 store.
            if constexpr (Branch<BR>::CONTIG) {
                int tau0 = tt*16 + grp*8;
                int seq = tau0 / Branch<BR>::L, j0 = tau0 - seq*Branch<BR>::L;
                bf16x8 v8 = { (__bf16)acc[0], (__bf16)acc[1], (__bf16)acc[2], (__bf16)acc[3],
                              (__bf16)acc[4], (__bf16)acc[5], (__bf16)acc[6], (__bf16)acc[7] };
                *(bf16x8*)(vT + ((size_t)(seq*HEADS_ + h)*HD_ + col)*Branch<BR>::Lp + j0) = v8;
            } else {
                #pragma unroll
                for (int r = 0; r < 8; ++r) {
                    int tau = tt*16 + r + grp*8;
                    int seq = tau / Branch<BR>::L, j = tau - seq*Branch<BR>::L;
                    vT[((size_t)(seq*HEADS_ + h)*HD_ + col)*Branch<BR>::Lp + j] = (__bf16)acc[r];
                }
            }
        } else {
            __bf16* dst = (sec == 0) ? qbuf : kbuf;
            #pragma unroll
            for (int r = 0; r < 8; ++r) {
                int tau = tt*16 + r + grp*8;
                dst[(size_t)tau*PC_ + h*HD_ + col] = (__bf16)acc[r];
            }
        }
    }
}

// --------------- phase 2: flash attention, 32 queries per wave ------------
template <int BR>
__global__ __launch_bounds__(32)
void attn_kernel(const __bf16* __restrict__ qbuf, const __bf16* __restrict__ kbuf,
                 const __bf16* __restrict__ vT, __bf16* __restrict__ ctx)
{
    constexpr int L  = Branch<BR>::L;
    constexpr int Lp = Branch<BR>::Lp;
    __shared__ __bf16 Pt[2][16][32];   // probabilities (natural key order)
    const int lane = threadIdx.x;
    const int row = lane & 15, grp = lane >> 4, col = lane & 15;
    const int h = blockIdx.y, seq = blockIdx.z;
    const int qbase = blockIdx.x * 32;

    // Q A-fragments: d=16 zero-padded to K=32 -> grp0 lanes hold d=0..15, grp1 zeros
    v16bf aq[2];
    #pragma unroll
    for (int qi = 0; qi < 2; ++qi) {
        int tau = seq*L + qbase + qi*16 + row;
        bf16x8 lo = {}, hi = {};
        if (grp == 0) {
            lo = *(const bf16x8*)(qbuf + (size_t)tau*PC_ + h*HD_);
            hi = *(const bf16x8*)(qbuf + (size_t)tau*PC_ + h*HD_ + 8);
        }
        aq[qi] = cat8(lo, hi);
    }

    float mrun[2][8], lrun[2][8];
    v8f oacc[2];
    #pragma unroll
    for (int qi = 0; qi < 2; ++qi)
        #pragma unroll
        for (int r = 0; r < 8; ++r) {
            mrun[qi][r] = -1e30f; lrun[qi][r] = 0.0f; oacc[qi][r] = 0.0f;
        }
    const float scale = 0.25f;   // 1/sqrt(d), d=16

    const __bf16* vbase = vT + ((size_t)(seq*HEADS_ + h)*HD_ + col)*Lp;
    const int nchunks = (L + 31) / 32;

    for (int c = 0; c < nchunks; ++c) {
        // K B-fragments (interleaved keys: S0 <-> key 2*col, S1 <-> 2*col+1);
        // only grp0 lanes carry d=0..15, grp1 is the K=16..31 zero padding.
        v16bf b0 = {}, b1 = {};
        if (grp == 0) {
            const __bf16* kr0 = kbuf + (size_t)(seq*L + c*32 + 2*col)*PC_ + h*HD_;
            const __bf16* kr1 = kr0 + PC_;
            b0 = cat8(*(const bf16x8*)kr0, *(const bf16x8*)(kr0 + 8));
            b1 = cat8(*(const bf16x8*)kr1, *(const bf16x8*)(kr1 + 8));
        }
        // V B-fragment: keys grp*8+0..7 and 16+grp*8+0..7 (V pre-transposed)
        const __bf16* vr = vbase + c*32 + grp*8;
        v16bf bv = cat8(*(const bf16x8*)vr, *(const bf16x8*)(vr + 16));

        __syncthreads();   // previous chunk's Pt fully consumed

        const int k0 = c*32 + 2*col, k1 = k0 + 1;
        #pragma unroll
        for (int qi = 0; qi < 2; ++qi) {
            v8f s0 = {}, s1 = {};
            s0 = wmma_bf16(aq[qi], b0, s0);
            s1 = wmma_bf16(aq[qi], b1, s1);
            #pragma unroll
            for (int r = 0; r < 8; ++r) {
                s0[r] = (k0 < L) ? s0[r]*scale : -1e30f;
                s1[r] = (k1 < L) ? s1[r]*scale : -1e30f;
            }
            // online softmax: a row's keys live in one 16-lane half
            #pragma unroll
            for (int r = 0; r < 8; ++r) {
                float mx = fmaxf(s0[r], s1[r]);
                #pragma unroll
                for (int msk = 1; msk < 16; msk <<= 1)
                    mx = fmaxf(mx, __shfl_xor(mx, msk, 32));
                float mnew = fmaxf(mrun[qi][r], mx);
                float alpha = __expf(mrun[qi][r] - mnew);
                mrun[qi][r] = mnew;
                float p0 = __expf(s0[r] - mnew);
                float p1 = __expf(s1[r] - mnew);
                s0[r] = p0; s1[r] = p1;
                float rs = p0 + p1;
                #pragma unroll
                for (int msk = 1; msk < 16; msk <<= 1)
                    rs += __shfl_xor(rs, msk, 32);
                lrun[qi][r] = lrun[qi][r]*alpha + rs;
                oacc[qi][r] *= alpha;
            }
            // packed b32 stores: keys (2*col, 2*col+1) are adjacent
            #pragma unroll
            for (int r = 0; r < 8; ++r) {
                bf16x2 pp = { (__bf16)s0[r], (__bf16)s1[r] };
                *(bf16x2*)&Pt[qi][r + grp*8][2*col] = pp;
            }
        }
        __syncthreads();
        #pragma unroll
        for (int qi = 0; qi < 2; ++qi) {
            bf16x8 lo = *(const bf16x8*)&Pt[qi][row][grp*8];
            bf16x8 hi = *(const bf16x8*)&Pt[qi][row][16 + grp*8];
            oacc[qi] = wmma_bf16(cat8(lo, hi), bv, oacc[qi]);
        }
    }

    // normalize + store context (rows past L masked)
    #pragma unroll
    for (int qi = 0; qi < 2; ++qi)
        #pragma unroll
        for (int r = 0; r < 8; ++r) {
            int j = qbase + qi*16 + r + grp*8;
            if (j < L) {
                int tau = seq*L + j;
                ctx[(size_t)tau*PC_ + h*HD_ + col] = (__bf16)(oacc[qi][r] / lrun[qi][r]);
            }
        }
}

// ------ phase 3: output projection, fused concat+transpose+shuffle --------
template <int BR>
__global__ __launch_bounds__(32)
void proj_kernel(const __bf16* __restrict__ ctx, const float* __restrict__ wproj,
                 const float* __restrict__ bproj, float* __restrict__ out)
{
    __shared__ __bf16 BtT[16][128];
    const int lane = threadIdx.x;
    const int row = lane & 15, grp = lane >> 4, col = lane & 15;
    const int nt = blockIdx.y;   // 0..7

    {
        const int k0 = lane * 4;
        float w[4][16];
        #pragma unroll
        for (int q = 0; q < 4; ++q) {
            const float4* src = (const float4*)(wproj + (size_t)(k0 + q) * PC_ + nt * 16);
            #pragma unroll
            for (int c4 = 0; c4 < 4; ++c4) {
                float4 v = src[c4];
                w[q][c4*4+0] = v.x; w[q][c4*4+1] = v.y;
                w[q][c4*4+2] = v.z; w[q][c4*4+3] = v.w;
            }
        }
        #pragma unroll
        for (int c = 0; c < 16; ++c) {
            bf16x4 p = { (__bf16)w[0][c], (__bf16)w[1][c],
                         (__bf16)w[2][c], (__bf16)w[3][c] };
            *(bf16x4*)&BtT[c][k0] = p;
        }
    }
    __syncthreads();

    v16bf bm[4];
    #pragma unroll
    for (int ks = 0; ks < 4; ++ks) {
        bf16x8 lo = *(const bf16x8*)&BtT[col][ks*32 + grp*8];
        bf16x8 hi = *(const bf16x8*)&BtT[col][ks*32 + 16 + grp*8];
        bm[ks] = cat8(lo, hi);
    }

    const float bias = bproj[nt*16 + col];
    const int co = (nt*16 + col)*4 + Branch<BR>::g;   // channel shuffle

    for (int it = 0; it < 8; ++it) {
        const int tt = blockIdx.x * 8 + it;
        const __bf16* arow = ctx + (size_t)(tt*16 + row)*PC_;

        v8f acc;
        #pragma unroll
        for (int r = 0; r < 8; ++r) acc[r] = bias;

        #pragma unroll
        for (int ks = 0; ks < 4; ++ks) {
            bf16x8 lo = *(const bf16x8*)(arow + ks*32 + grp*8);
            bf16x8 hi = *(const bf16x8*)(arow + ks*32 + 16 + grp*8);
            acc = wmma_bf16(cat8(lo, hi), bm[ks], acc);
        }

        if constexpr (Branch<BR>::CONTIG) {
            // lane's 8 rows are 8 consecutive t positions -> two b128 stores
            int tau0 = tt*16 + grp*8;
            int b, n, t; Branch<BR>::tok2bnt(tau0, b, n, t);
            float* o = out + (((size_t)b*C_ + co)*N_ + n)*T_ + t;
            float4 v0 = { acc[0], acc[1], acc[2], acc[3] };
            float4 v1 = { acc[4], acc[5], acc[6], acc[7] };
            *(float4*)(o)     = v0;
            *(float4*)(o + 4) = v1;
        } else {
            #pragma unroll
            for (int r = 0; r < 8; ++r) {
                int tau = tt*16 + r + grp*8;
                int b, n, t; Branch<BR>::tok2bnt(tau, b, n, t);
                out[(((size_t)b*C_ + co)*N_ + n)*T_ + t] = acc[r];
            }
        }
    }
}

// ----------------------- identity chunk (g = 0) ---------------------------
__global__ __launch_bounds__(256)
void ident_kernel(const float* __restrict__ x, float* __restrict__ out)
{
    int idx = blockIdx.x * blockDim.x + threadIdx.x;   // over B*N*T*PC/4
    if (idx >= B_ * N_ * T_ * PC_ / 4) return;
    int p4  = idx & 31;            // 32 float4 per token row
    int bnt = idx >> 5;
    int t = bnt & 63;
    int bn = bnt >> 6;
    int n = bn % 25, b = bn / 25;
    float4 v = *(const float4*)(x + (size_t)bnt * C_ + p4 * 4);
    float* o = out + (((size_t)b*C_ + p4*16)*N_ + n)*T_ + t;   // co = 4*p
    const size_t cs = (size_t)N_ * T_ * 4;                     // 4 channels per p
    o[0] = v.x; o[cs] = v.y; o[2*cs] = v.z; o[3*cs] = v.w;
}

// ---------------------------------------------------------------------------
extern "C" void kernel_launch(void* const* d_in, const int* in_sizes, int n_in,
                              void* d_out, int out_size, void* d_ws, size_t ws_size,
                              hipStream_t stream)
{
    (void)in_sizes; (void)n_in; (void)out_size; (void)ws_size;
    const float* x = (const float*)d_in[0];
    const float* w[3][4];
    for (int br = 0; br < 3; ++br)
        for (int k = 0; k < 4; ++k)
            w[br][k] = (const float*)d_in[1 + br * 4 + k];

    float* out = (float*)d_out;
    __bf16* ws = (__bf16*)d_ws;

    const size_t SZQ = (size_t)TOKENS_ * PC_ + 8192;        // q / k / ctx
    const size_t SZV = (size_t)512 * PC_ * 32 + 8192;       // max SEQS*128*Lp
    const size_t STRIDE = 3 * SZQ + SZV;

    dim3 blk(32);
    const int gtiles = TOKENS_ / 128;   // 100 blocks, 8 token-tiles each

    // branch 0: 't'
    {
        __bf16 *q = ws, *k = q + SZQ, *c = k + SZQ, *v = c + SZQ;
        qkv_kernel<0><<<dim3(gtiles, 24), blk, 0, stream>>>(x, w[0][0], w[0][1], q, k, v);
        attn_kernel<0><<<dim3(Branch<0>::QB, HEADS_, Branch<0>::SEQS), blk, 0, stream>>>(q, k, v, c);
        proj_kernel<0><<<dim3(gtiles, 8), blk, 0, stream>>>(c, w[0][2], w[0][3], out);
    }
    // branch 1: 's'
    {
        __bf16 *base = ws + STRIDE;
        __bf16 *q = base, *k = q + SZQ, *c = k + SZQ, *v = c + SZQ;
        qkv_kernel<1><<<dim3(gtiles, 24), blk, 0, stream>>>(x, w[1][0], w[1][1], q, k, v);
        attn_kernel<1><<<dim3(Branch<1>::QB, HEADS_, Branch<1>::SEQS), blk, 0, stream>>>(q, k, v, c);
        proj_kernel<1><<<dim3(gtiles, 8), blk, 0, stream>>>(c, w[1][2], w[1][3], out);
    }
    // branch 2: 'st'
    {
        __bf16 *base = ws + 2 * STRIDE;
        __bf16 *q = base, *k = q + SZQ, *c = k + SZQ, *v = c + SZQ;
        qkv_kernel<2><<<dim3(gtiles, 24), blk, 0, stream>>>(x, w[2][0], w[2][1], q, k, v);
        attn_kernel<2><<<dim3(Branch<2>::QB, HEADS_, Branch<2>::SEQS), blk, 0, stream>>>(q, k, v, c);
        proj_kernel<2><<<dim3(gtiles, 8), blk, 0, stream>>>(c, w[2][2], w[2][3], out);
    }
    // identity chunk (vectorized)
    {
        int total = B_ * N_ * T_ * PC_ / 4;
        ident_kernel<<<(total + 255) / 256, 256, 0, stream>>>(x, out);
    }
}
